// SignMaskLinear_12266426597796
// MI455X (gfx1250) — compile-verified, matrix operand order checked by reading, not run on hardware
//
#include <hip/hip_runtime.h>

// out[32768,1024] = x[32768,1024] @ sign(W)[1024,1024]^T + b
#define TOKENS 32768
#define KDIM   1024
#define NDIM   1024

#define BM 128
#define BN 128
#define BK 64
#define LDST 72            // padded LDS row stride in bf16 elems (144B, 16B aligned)
#define NKT (KDIM / BK)    // 16 K-tiles

typedef __attribute__((ext_vector_type(16))) __bf16 v16bf;
typedef __attribute__((ext_vector_type(8)))  __bf16 v8bf;
typedef __attribute__((ext_vector_type(8)))  float  v8f;

// ---------------- CDNA5 async global->LDS copy (ASYNCcnt path) ----------------
#if defined(__AMDGCN__) && __has_builtin(__builtin_amdgcn_global_load_async_to_lds_b128)
#define USE_ASYNC_LDS 1
#endif

#ifdef USE_ASYNC_LDS
// Builtin signature (verified via hipcc diagnostics): (v4i AS1*, v4i AS3*, imm offset, imm cpol)
typedef int v4i_ __attribute__((vector_size(16)));
typedef __attribute__((address_space(1))) v4i_ AS1v4i;
typedef __attribute__((address_space(3))) v4i_ AS3v4i;

__device__ __forceinline__ void async_copy_b128(const unsigned short* g, unsigned short* l) {
  // generic->AS1: same 64-bit value; generic->AS3: LDS offset = low 32 bits (aperture rule)
  __builtin_amdgcn_global_load_async_to_lds_b128(
      (AS1v4i*)(unsigned long long)(const void*)g,
      (AS3v4i*)(unsigned int)(unsigned long long)(void*)l,
      /*offset=*/0, /*cpol=*/0);
}
__device__ __forceinline__ void wait_async0() {
#if __has_builtin(__builtin_amdgcn_s_wait_asynccnt)
  __builtin_amdgcn_s_wait_asynccnt(0);
#else
  asm volatile("s_wait_asynccnt 0" ::: "memory");
#endif
}
#endif

__device__ __forceinline__ unsigned int f2bf_rne(float f) {
  unsigned int u = __float_as_uint(f);
  u += 0x7FFFu + ((u >> 16) & 1u);   // round-to-nearest-even
  return (u >> 16) & 0xFFFFu;
}

// ------------------------------------------------------------------
// Kernel 1: ternary-quantize f32 weight -> bf16 {-1,0,+1} (exact in bf16)
// ------------------------------------------------------------------
__global__ __launch_bounds__(256)
void ternary_quant_kernel(const float* __restrict__ w,
                          unsigned short* __restrict__ wq) {
  const int base = (blockIdx.x * 256 + threadIdx.x) * 8;
  const float4 a = *(const float4*)(w + base);
  const float4 b = *(const float4*)(w + base + 4);
  const float f[8] = {a.x, a.y, a.z, a.w, b.x, b.y, b.z, b.w};
  unsigned int p[4];
#pragma unroll
  for (int i = 0; i < 4; ++i) {
    const float f0 = f[2 * i], f1 = f[2 * i + 1];
    const unsigned int q0 = (f0 > 0.f) ? 0x3F80u : ((f0 < 0.f) ? 0xBF80u : 0u);
    const unsigned int q1 = (f1 > 0.f) ? 0x3F80u : ((f1 < 0.f) ? 0xBF80u : 0u);
    p[i] = q0 | (q1 << 16);
  }
  uint4 o; o.x = p[0]; o.y = p[1]; o.z = p[2]; o.w = p[3];
  *(uint4*)(wq + base) = o;
}

// ------------------------------------------------------------------
// Kernel 2: double-buffered bf16 WMMA GEMM.
// Block = 128x128 tile, 8 wave32s, each wave = 32x64 (2x4 WMMA accs).
// W tile = 128 rows x 64 halves = 1024 x 16B chunks -> 4 chunks/thread.
// ------------------------------------------------------------------
__global__ __launch_bounds__(256, 2)
void ternary_gemm_wmma_kernel(const float* __restrict__ x,
                              const unsigned short* __restrict__ wq,
                              const float* __restrict__ bias,
                              float* __restrict__ out) {
  __shared__ unsigned short Xs[2][BM * LDST];
  __shared__ unsigned short Ws[2][BN * LDST];

  const int tid  = threadIdx.x;
  const int lane = tid & 31;
  const int wave = tid >> 5;
  const int wm   = wave & 3;    // M sub-tile (32 rows)
  const int wn   = wave >> 2;   // N sub-tile (64 cols)
  const int l15  = lane & 15;
  const int lhi  = lane >> 4;   // lane half
  const int m0   = blockIdx.y * BM;
  const int n0   = blockIdx.x * BN;

  const int xrow = tid >> 4;          // 0..15
  const int xcol = (tid & 15) * 4;    // float4 column

  v8f acc[2][4];
#pragma unroll
  for (int i = 0; i < 2; ++i)
#pragma unroll
    for (int j = 0; j < 4; ++j)
#pragma unroll
      for (int r = 0; r < 8; ++r) acc[i][j][r] = 0.f;

  float4 xr[8];
#ifndef USE_ASYNC_LDS
  uint4 wr[4];
#endif

  // -------- prologue: stage K-tile 0 into buffer 0 --------
#pragma unroll
  for (int it = 0; it < 8; ++it)
    xr[it] = *(const float4*)(x + (size_t)(m0 + xrow + it * 16) * KDIM + xcol);
#pragma unroll
  for (int it = 0; it < 4; ++it) {
    const int c = tid + it * 256, row = c >> 3, col = (c & 7) * 8;
    const unsigned short* g = wq + (size_t)(n0 + row) * KDIM + col;
#ifdef USE_ASYNC_LDS
    async_copy_b128(g, &Ws[0][row * LDST + col]);
#else
    wr[it] = *(const uint4*)g;
#endif
  }
#pragma unroll
  for (int it = 0; it < 8; ++it) {
    const int row = xrow + it * 16;
    const float4 v = xr[it];
    uint2 pk;
    pk.x = f2bf_rne(v.x) | (f2bf_rne(v.y) << 16);
    pk.y = f2bf_rne(v.z) | (f2bf_rne(v.w) << 16);
    *(uint2*)(&Xs[0][row * LDST + xcol]) = pk;
  }
#ifdef USE_ASYNC_LDS
  wait_async0();
#else
#pragma unroll
  for (int it = 0; it < 4; ++it) {
    const int c = tid + it * 256, row = c >> 3, col = (c & 7) * 8;
    *(uint4*)(&Ws[0][row * LDST + col]) = wr[it];
  }
#endif
  __syncthreads();

  // -------- main pipelined K loop --------
  for (int kb = 0; kb < NKT; ++kb) {
    const int cur = kb & 1, nxt = cur ^ 1;
    const bool have_next = (kb + 1) < NKT;
    const int k1 = (kb + 1) * BK;

    // issue next tile's global loads before computing on current tile
    if (have_next) {
#pragma unroll
      for (int it = 0; it < 8; ++it)
        xr[it] = *(const float4*)(x + (size_t)(m0 + xrow + it * 16) * KDIM + k1 + xcol);
#pragma unroll
      for (int it = 0; it < 4; ++it) {
        const int c = tid + it * 256, row = c >> 3, col = (c & 7) * 8;
        const unsigned short* g = wq + (size_t)(n0 + row) * KDIM + k1 + col;
#ifdef USE_ASYNC_LDS
        async_copy_b128(g, &Ws[nxt][row * LDST + col]);
#else
        wr[it] = *(const uint4*)g;
#endif
      }
      if (kb + 2 < NKT)  // speculative prefetch two tiles ahead
        __builtin_prefetch(x + (size_t)(m0 + xrow) * KDIM + k1 + BK + xcol, 0, 1);
    }

    // ---- compute on buffer `cur`: two K=32 WMMA steps ----
    const unsigned short* xsc = &Xs[cur][0];
    const unsigned short* wsc = &Ws[cur][0];
#pragma unroll
    for (int ks = 0; ks < 2; ++ks) {
      // 16-bit A layout: lanes 0-15 hold K {0..7,16..23}, lanes 16-31 K {8..15,24..31}
      const int kf = ks * 32 + lhi * 8;

      v16bf afrag[2];
#pragma unroll
      for (int i = 0; i < 2; ++i) {
        const unsigned short* p = xsc + (wm * 32 + i * 16 + l15) * LDST + kf;
        const v8bf lo = *(const v8bf*)p;
        const v8bf hi = *(const v8bf*)(p + 16);
#pragma unroll
        for (int u = 0; u < 8; ++u) { afrag[i][u] = lo[u]; afrag[i][u + 8] = hi[u]; }
      }

      v16bf bfrag[4];
#pragma unroll
      for (int j = 0; j < 4; ++j) {
        const unsigned short* p = wsc + (wn * 64 + j * 16 + l15) * LDST + kf;
        const v8bf lo = *(const v8bf*)p;
        const v8bf hi = *(const v8bf*)(p + 16);
#pragma unroll
        for (int u = 0; u < 8; ++u) { bfrag[j][u] = lo[u]; bfrag[j][u + 8] = hi[u]; }
      }

#pragma unroll
      for (int i = 0; i < 2; ++i)
#pragma unroll
        for (int j = 0; j < 4; ++j)
          acc[i][j] = __builtin_amdgcn_wmma_f32_16x16x32_bf16(
              /*neg_a=*/false, afrag[i],
              /*neg_b=*/false, bfrag[j],
              /*c_mod=*/(short)0, acc[i][j],
              /*reuse_a=*/false, /*reuse_b=*/false);
    }

    // convert + stage next X tile (and W fallback) into the other buffer
    if (have_next) {
#pragma unroll
      for (int it = 0; it < 8; ++it) {
        const int row = xrow + it * 16;
        const float4 v = xr[it];
        uint2 pk;
        pk.x = f2bf_rne(v.x) | (f2bf_rne(v.y) << 16);
        pk.y = f2bf_rne(v.z) | (f2bf_rne(v.w) << 16);
        *(uint2*)(&Xs[nxt][row * LDST + xcol]) = pk;
      }
#ifndef USE_ASYNC_LDS
#pragma unroll
      for (int it = 0; it < 4; ++it) {
        const int c = tid + it * 256, row = c >> 3, col = (c & 7) * 8;
        *(uint4*)(&Ws[nxt][row * LDST + col]) = wr[it];
      }
#endif
    }
#ifdef USE_ASYNC_LDS
    wait_async0();   // own async W-copies visible in LDS before the barrier
#endif
    __syncthreads();
  }

  // -------- epilogue: C/D layout -> row-major + bias --------
#pragma unroll
  for (int j = 0; j < 4; ++j) {
    const int n = n0 + wn * 64 + j * 16 + l15;
    const float bv = bias[n];
#pragma unroll
    for (int i = 0; i < 2; ++i) {
      const int mrow = m0 + wm * 32 + i * 16 + lhi * 8;   // VGPR r: M=r (+8 hi half)
#pragma unroll
      for (int r = 0; r < 8; ++r)
        out[(size_t)(mrow + r) * NDIM + n] = acc[i][j][r] + bv;
    }
  }
}

// ------------------------------------------------------------------
extern "C" void kernel_launch(void* const* d_in, const int* in_sizes, int n_in,
                              void* d_out, int out_size, void* d_ws, size_t ws_size,
                              hipStream_t stream) {
  const float* x    = (const float*)d_in[0];   // [32768, 1024]
  const float* w    = (const float*)d_in[1];   // [1024, 1024]
  const float* bias = (const float*)d_in[2];   // [1024]
  float* out = (float*)d_out;
  unsigned short* wq = (unsigned short*)d_ws;  // 1024*1024 bf16 = 2 MiB scratch

  ternary_quant_kernel<<<(NDIM * KDIM) / (256 * 8), 256, 0, stream>>>(w, wq);

  dim3 grid(NDIM / BN, TOKENS / BM);           // x = N tiles (L2 reuse of x), y = M tiles
  ternary_gemm_wmma_kernel<<<grid, 256, 0, stream>>>(x, wq, bias, out);
}
